// RNNCellFCModel_54443005444542
// MI455X (gfx1250) — matrix operand order repaired and checked by reading, not compiled, not run on hardware
//
#include <hip/hip_runtime.h>
#include <hip/hip_bf16.h>
#include <initializer_list>
#include <stdint.h>

// ---------------------------------------------------------------------------
// Types for WMMA / TDM operands (gfx1250, wave32)
// ---------------------------------------------------------------------------
typedef __attribute__((ext_vector_type(16))) __bf16          bf16x16;
typedef __attribute__((ext_vector_type(8)))  float           f32x8;
typedef __attribute__((ext_vector_type(8)))  unsigned short  u16x8;
typedef __attribute__((ext_vector_type(4)))  unsigned int    u32x4;
typedef __attribute__((ext_vector_type(4)))  int             i32x4;
typedef __attribute__((ext_vector_type(8)))  int             i32x8;

union BfTile {
  bf16x16 v;
  u16x8   u[2];
};

#define KC    64   // K elements staged per chunk (2 WMMA K-steps)
#define PITCH 72   // LDS row pitch in elements: 64 data + 8 pad (144B, bank spread)

#ifdef __has_builtin
#  if __has_builtin(__builtin_amdgcn_tensor_load_to_lds)
#    define USE_TDM 1
#  endif
#endif
#ifndef USE_TDM
#  define USE_TDM 0
#endif

__device__ __forceinline__ unsigned short f32_to_bf16_rne(float f) {
  unsigned int u = __float_as_uint(f);
  unsigned int r = u + 0x7FFFu + ((u >> 16) & 1u);   // round-to-nearest-even
  return (unsigned short)(r >> 16);
}

// WGP-scope prefetch: pull into all cache levels, stop on WGP$ hit (scope 0).
// __builtin_prefetch only reaches SE/SYS scopes; syntax below matches the
// compiler's own global_prefetch_b8 output.
__device__ __forceinline__ void prefetch_wgp(const void* p) {
#if defined(__gfx1250__)
  asm volatile("global_prefetch_b8 %0, off" :: "v"(p));
#else
  __builtin_prefetch(p, 0, 3);
#endif
}

#if USE_TDM
// Issue one TDM 2D tile load: 64 rows x KC bf16 elements from global (row
// stride zs elements) into LDS at ldst, padded to PITCH elements per row via
// the descriptor's pad feature (4 DWORDs pad per 32 DWORDs stored).
// D# layout per CDNA5 ISA section 8 (group0 128b, group1 256b).
__device__ __forceinline__ void tdm_stage(const unsigned short* gsrc,
                                          long long zs_elems,
                                          unsigned short* ldst) {
  unsigned long long ga = (unsigned long long)gsrc;
  unsigned lds_off = (unsigned)(size_t)ldst;    // generic->LDS: low 32 bits
  u32x4 g0;
  g0[0] = 1u;                                   // count=1, user, no gather
  g0[1] = lds_off;                              // lds_addr
  g0[2] = (unsigned)ga;                         // global_addr[31:0]
  g0[3] = (unsigned)((ga >> 32) & 0x01FFFFFFu)  // global_addr[56:32]
        | (2u << 30);                           // type = 2 ("image")
  unsigned long long st = (unsigned long long)zs_elems;   // dim0 stride (elems)
  i32x8 g1;
  g1[0] = (int)((1u << 16)        // data_size = 1 (2 bytes)
              | (1u << 20)        // pad_enable
              | (4u << 22)        // pad_interval code 4 -> every 32 DWORDs
              | (3u << 25));      // pad_amount  code 3 -> 4 DWORDs
  g1[1] = (int)((unsigned)KC << 16);            // tensor_dim0[15:0] (=KC)
  g1[2] = (int)(64u << 16);                     // tensor_dim0 hi=0 | tensor_dim1=64
  g1[3] = (int)((unsigned)KC << 16);            // tensor_dim1 hi=0 | tile_dim0=KC
  g1[4] = (int)(64u);                           // tile_dim1=64 rows, tile_dim2=0
  g1[5] = (int)(unsigned)(st & 0xFFFFFFFFu);    // tensor_dim0_stride[31:0]
  g1[6] = (int)(unsigned)((st >> 32) & 0xFFFFu);// stride[47:32] | dim1_stride lo=0
  g1[7] = 0;
  i32x4 gz = {0, 0, 0, 0};                      // 2D tensor: groups 2/3 unused
#if __clang_major__ >= 23
  i32x8 gz8 = {0, 0, 0, 0, 0, 0, 0, 0};
  __builtin_amdgcn_tensor_load_to_lds(g0, g1, gz, gz, gz8, 0);
#else
  __builtin_amdgcn_tensor_load_to_lds(g0, g1, gz, gz, 0);
#endif
}
#endif  // USE_TDM

// ---------------------------------------------------------------------------
// fp32 -> bf16 conversion (one-time: weights / inputs / initial state)
// ---------------------------------------------------------------------------
__global__ void f32_to_bf16_kernel(const float* __restrict__ in,
                                   unsigned short* __restrict__ out,
                                   long long n) {
  long long i = (long long)blockIdx.x * blockDim.x + threadIdx.x;
  long long stride = (long long)gridDim.x * blockDim.x;
  for (; i < n; i += stride) out[i] = f32_to_bf16_rne(in[i]);
}

__global__ void bias_sum_kernel(const float* __restrict__ a,
                                const float* __restrict__ b,
                                float* __restrict__ o, int n) {
  int i = blockIdx.x * blockDim.x + threadIdx.x;
  if (i < n) o[i] = a[i] + b[i];
}

// ---------------------------------------------------------------------------
// Segmented GEMM:   out[64, N] = act( sum_s Z_s[64, Ks] * W_s[N, Ks]^T + bias )
// ---------------------------------------------------------------------------
struct Seg {
  const unsigned short* z;   // bf16 activations, row b at z + b*zs
  const unsigned short* w;   // bf16 weights, output-row n at w + n*ws (cols = K)
  long long zs;
  long long ws;
  int width;                 // K for this segment; multiple of KC
  int _pad;
};

struct GemmArgs {
  Seg seg[4];
  const float*    bias;
  unsigned short* out_bf;    // bf16 hidden state [64, N] (row stride N), or null
  float*          out_f32;   // fp32 output, row b at out_f32 + b*out_stride, or null
  long long       out_stride;
  int nseg;
  int N;
  int act_tanh;
  int _pad;
};

// 4 waves / block; wave w owns the 64x32 output tile at columns
// [blockIdx.x*128 + w*32, +32). Activation chunks (64 x KC) are DMA'd once per
// block into LDS by the Tensor Data Mover (double buffered, TENSORcnt), shared
// by all 4 waves -> 4x less L2 activation traffic. Weight tiles stream
// directly global->VGPR (no intra-block reuse).
__global__ __launch_bounds__(128)
void rnn_gemm_wmma(GemmArgs A) {
  __shared__ __align__(16) unsigned short zbuf[2][64 * PITCH];

  const int tid   = threadIdx.x;
  const int lane  = tid & 31;
  const int wave  = tid >> 5;
  const int n0    = blockIdx.x * 128 + wave * 32;
  const int lrow  = lane & 15;   // A: M within tile; B/D: N within tile
  const int khalf = lane >> 4;   // which K half this lane holds (ISA 7.12.2)

  // Leader predicate as a *scalar* (readfirstlane -> SCC branch, not EXEC
  // predication): TDM issue ignores EXEC, so it must be branched around on
  // non-leader waves or every wave would issue a duplicate DMA.
  const bool leader = (__builtin_amdgcn_readfirstlane(wave) == 0);

  const f32x8 zero = {0.f, 0.f, 0.f, 0.f, 0.f, 0.f, 0.f, 0.f};
  f32x8 acc[2][4];
#pragma unroll
  for (int nt = 0; nt < 2; ++nt)
#pragma unroll
    for (int m = 0; m < 4; ++m) acc[nt][m] = zero;

  for (int s = 0; s < A.nseg; ++s) {
    const unsigned short* zbase = A.seg[s].z;
    const unsigned short* wbase = A.seg[s].w;
    const long long zs = A.seg[s].zs;
    const long long ws = A.seg[s].ws;
    const int nchunk   = A.seg[s].width / KC;

    // B (32x16 bf16): lane holds column n0+lrow, K=[0..7]/[16..23] offset by 8*khalf.
    const unsigned short* wrow0 = wbase + (long long)(n0 + lrow) * ws + (khalf << 3);
    const unsigned short* wrow1 = wrow0 + 16ll * ws;   // second N-subtile

#if USE_TDM
    if (leader) tdm_stage(zbase, zs, &zbuf[0][0]);             // prologue chunk 0
#endif
    for (int c = 0; c < nchunk; ++c) {
#if USE_TDM
      if (leader) __builtin_amdgcn_s_wait_tensorcnt(0);        // chunk c landed
      __syncthreads();                                         // all waves: LDS valid,
                                                               // buf[(c+1)&1] free
      if (leader && c + 1 < nchunk)
        tdm_stage(zbase + (long long)(c + 1) * KC, zs, &zbuf[(c + 1) & 1][0]);
      const unsigned short* lz0 = &zbuf[c & 1][0];
#else
      // Fallback: cooperative synchronous staging, single buffer.
      __syncthreads();
      {
        const int crow = tid >> 1, coff = (tid & 1) * 32;      // 128 thr: half-row each
        const unsigned short* src = zbase + (long long)crow * zs + (long long)c * KC + coff;
        unsigned short* dst = &zbuf[0][(size_t)crow * PITCH + coff];
#pragma unroll
        for (int q = 0; q < 4; ++q)
          *(u16x8*)(dst + q * 8) = *(const u16x8*)(src + q * 8);
      }
      __syncthreads();
      const unsigned short* lz0 = &zbuf[0][0];
#endif

      // Pull the weight chunk two ahead (c+2) into the WGP caches (global_prefetch_b8).
      const int kbase = c * KC;
      prefetch_wgp(wrow0 + kbase + 2 * KC);
      prefetch_wgp(wrow1 + kbase + 2 * KC);

#pragma unroll
      for (int kk = 0; kk < KC; kk += 32) {
        const int k0 = kbase + kk;
        BfTile a[4], b0, b1;
        // A (16x32 bf16) from LDS: lanes 0-15 hold M=lrow, K=kk+[0..7]/[16..23];
        // lanes 16-31 hold K=kk+[8..15]/[24..31]. Row pitch 144B.
        const unsigned short* lz = lz0 + (size_t)lrow * PITCH + kk + (khalf << 3);
#pragma unroll
        for (int m = 0; m < 4; ++m) {
          a[m].u[0] = *(const u16x8*)(lz + (size_t)(m * 16) * PITCH);
          a[m].u[1] = *(const u16x8*)(lz + (size_t)(m * 16) * PITCH + 16);
        }
        b0.u[0] = *(const u16x8*)(wrow0 + k0);
        b0.u[1] = *(const u16x8*)(wrow0 + k0 + 16);
        b1.u[0] = *(const u16x8*)(wrow1 + k0);
        b1.u[1] = *(const u16x8*)(wrow1 + k0 + 16);
#pragma unroll
        for (int m = 0; m < 4; ++m) {
          acc[0][m] = __builtin_amdgcn_wmma_f32_16x16x32_bf16(
              false, a[m].v, false, b0.v, (short)0, acc[0][m], false, false);
          acc[1][m] = __builtin_amdgcn_wmma_f32_16x16x32_bf16(
              false, a[m].v, false, b1.v, (short)0, acc[1][m], false, false);
        }
      }
    }
  }

  // D layout (ISA 7.12.2): lane -> N = lrow, VGPR r -> M = r + khalf*8.
#pragma unroll
  for (int nt = 0; nt < 2; ++nt) {
    const int col = n0 + nt * 16 + lrow;
    const float bv = A.bias ? A.bias[col] : 0.0f;
#pragma unroll
    for (int m = 0; m < 4; ++m) {
#pragma unroll
      for (int r = 0; r < 8; ++r) {
        const int row = m * 16 + khalf * 8 + r;
        float v = acc[nt][m][r] + bv;
        if (A.act_tanh) v = tanhf(v);
        if (A.out_bf)
          A.out_bf[(long long)row * A.N + col] = f32_to_bf16_rne(v);
        if (A.out_f32)
          A.out_f32[(long long)row * A.out_stride + col] = v;
      }
    }
  }
}

// ---------------------------------------------------------------------------
// Host orchestration
// ---------------------------------------------------------------------------
extern "C" void kernel_launch(void* const* d_in, const int* in_sizes, int n_in,
                              void* d_out, int out_size, void* d_ws, size_t ws_size,
                              hipStream_t stream) {
  (void)in_sizes; (void)n_in; (void)out_size;

  const int B = 64, T = 128, XON = 1024, XOFF = 1024, IN = 2048;
  const int H4E = 2048, H4I = 512, H23E = 2048, H23I = 512;
  const int O4E = 4096, O4I = 1024, O23E = 4096, O23I = 1024;

  // d_in order = setup_inputs() dict order (depth-first through 'params').
  const float* x_on   = (const float*)d_in[0];
  const float* x_off  = (const float*)d_in[1];
  const float* h4e_in = (const float*)d_in[2];
  const float* h4i_in = (const float*)d_in[3];
  const float* h23e_in= (const float*)d_in[4];
  const float* h23i_in= (const float*)d_in[5];
  const float* W4e_ih = (const float*)d_in[6];   // [2048, 2560]
  const float* W4e_hh = (const float*)d_in[7];   // [2048, 2048]
  const float* b4e_ih = (const float*)d_in[8];
  const float* b4e_hh = (const float*)d_in[9];
  const float* W4i_ih = (const float*)d_in[10];  // [512, 4096]
  const float* W4i_hh = (const float*)d_in[11];  // [512, 512]
  const float* b4i_ih = (const float*)d_in[12];
  const float* b4i_hh = (const float*)d_in[13];
  const float* W23e_ih= (const float*)d_in[14];  // [2048, 3072]
  const float* W23e_hh= (const float*)d_in[15];  // [2048, 2048]
  const float* b23e_ih= (const float*)d_in[16];
  const float* b23e_hh= (const float*)d_in[17];
  const float* W23i_ih= (const float*)d_in[18];  // [512, 4608]
  const float* W23i_hh= (const float*)d_in[19];  // [512, 512]
  const float* b23i_ih= (const float*)d_in[20];
  const float* b23i_hh= (const float*)d_in[21];
  const float* Wfc4e  = (const float*)d_in[22];  // [4096, 2048]
  const float* bfc4e  = (const float*)d_in[23];
  const float* Wfc4i  = (const float*)d_in[24];  // [1024, 512]
  const float* bfc4i  = (const float*)d_in[25];
  const float* Wfc23e = (const float*)d_in[26];  // [4096, 2048]
  const float* bfc23e = (const float*)d_in[27];
  const float* Wfc23i = (const float*)d_in[28];  // [1024, 512]
  const float* bfc23i = (const float*)d_in[29];

  // Workspace bump allocator: bf16 weights (~85MB) fit the 192MB L2, halving
  // per-step weight traffic vs fp32 and feeding the bf16 WMMA fast path.
  char* ws = (char*)d_ws;
  size_t off = 0;
  auto alloc_us = [&](size_t n) -> unsigned short* {
    unsigned short* p = (unsigned short*)(ws + off);
    off += ((n * 2 + 255) & ~(size_t)255);
    return p;
  };
  auto alloc_f = [&](size_t n) -> float* {
    float* p = (float*)(ws + off);
    off += ((n * 4 + 255) & ~(size_t)255);
    return p;
  };

  unsigned short* xon_bf  = alloc_us((size_t)B * T * XON);
  unsigned short* xoff_bf = alloc_us((size_t)B * T * XOFF);
  unsigned short* w4e_ih  = alloc_us((size_t)H4E * (IN + H4I));
  unsigned short* w4e_hh  = alloc_us((size_t)H4E * H4E);
  unsigned short* w4i_ih  = alloc_us((size_t)H4I * (IN + H4E));
  unsigned short* w4i_hh  = alloc_us((size_t)H4I * H4I);
  unsigned short* w23e_ih = alloc_us((size_t)H23E * (H4E + H4I + H23I));
  unsigned short* w23e_hh = alloc_us((size_t)H23E * H23E);
  unsigned short* w23i_ih = alloc_us((size_t)H23I * (H4E + H4I + H23E));
  unsigned short* w23i_hh = alloc_us((size_t)H23I * H23I);
  unsigned short* wfc4e   = alloc_us((size_t)O4E * H4E);
  unsigned short* wfc4i   = alloc_us((size_t)O4I * H4I);
  unsigned short* wfc23e  = alloc_us((size_t)O23E * H23E);
  unsigned short* wfc23i  = alloc_us((size_t)O23I * H23I);
  float* bs4e  = alloc_f(H4E);
  float* bs4i  = alloc_f(H4I);
  float* bs23e = alloc_f(H23E);
  float* bs23i = alloc_f(H23I);
  unsigned short* hb4e[2]  = {alloc_us((size_t)B * H4E),  alloc_us((size_t)B * H4E)};
  unsigned short* hb4i[2]  = {alloc_us((size_t)B * H4I),  alloc_us((size_t)B * H4I)};
  unsigned short* hb23e[2] = {alloc_us((size_t)B * H23E), alloc_us((size_t)B * H23E)};
  unsigned short* hb23i[2] = {alloc_us((size_t)B * H23I), alloc_us((size_t)B * H23I)};
  if (off > ws_size) return;  // workspace too small; refuse to scribble OOB

  auto conv = [&](const float* src, unsigned short* dst, long long n) {
    long long g = (n + 255) / 256; if (g > 2048) g = 2048;
    f32_to_bf16_kernel<<<dim3((unsigned)g), dim3(256), 0, stream>>>(src, dst, n);
  };
  conv(x_on,  xon_bf,  (long long)B * T * XON);
  conv(x_off, xoff_bf, (long long)B * T * XOFF);
  conv(W4e_ih,  w4e_ih,  (long long)H4E * (IN + H4I));
  conv(W4e_hh,  w4e_hh,  (long long)H4E * H4E);
  conv(W4i_ih,  w4i_ih,  (long long)H4I * (IN + H4E));
  conv(W4i_hh,  w4i_hh,  (long long)H4I * H4I);
  conv(W23e_ih, w23e_ih, (long long)H23E * (H4E + H4I + H23I));
  conv(W23e_hh, w23e_hh, (long long)H23E * H23E);
  conv(W23i_ih, w23i_ih, (long long)H23I * (H4E + H4I + H23E));
  conv(W23i_hh, w23i_hh, (long long)H23I * H23I);
  conv(Wfc4e,  wfc4e,  (long long)O4E * H4E);
  conv(Wfc4i,  wfc4i,  (long long)O4I * H4I);
  conv(Wfc23e, wfc23e, (long long)O23E * H23E);
  conv(Wfc23i, wfc23i, (long long)O23I * H23I);
  conv(h4e_in,  hb4e[0],  (long long)B * H4E);
  conv(h4i_in,  hb4i[0],  (long long)B * H4I);
  conv(h23e_in, hb23e[0], (long long)B * H23E);
  conv(h23i_in, hb23i[0], (long long)B * H23I);
  bias_sum_kernel<<<(H4E + 255) / 256, 256, 0, stream>>>(b4e_ih,  b4e_hh,  bs4e,  H4E);
  bias_sum_kernel<<<(H4I + 255) / 256, 256, 0, stream>>>(b4i_ih,  b4i_hh,  bs4i,  H4I);
  bias_sum_kernel<<<(H23E + 255) / 256, 256, 0, stream>>>(b23e_ih, b23e_hh, bs23e, H23E);
  bias_sum_kernel<<<(H23I + 255) / 256, 256, 0, stream>>>(b23i_ih, b23i_hh, bs23i, H23I);

  auto S = [](const unsigned short* z, long long zs,
              const unsigned short* w, long long wsd, int width) {
    Seg s; s.z = z; s.w = w; s.zs = zs; s.ws = wsd; s.width = width; s._pad = 0;
    return s;
  };
  auto gemm = [&](int N, std::initializer_list<Seg> segs, const float* bias,
                  unsigned short* out_bf, float* out_f32, long long ostride, int act) {
    GemmArgs a{};
    int i = 0;
    for (const Seg& s : segs) a.seg[i++] = s;
    a.nseg = i; a.N = N; a.bias = bias;
    a.out_bf = out_bf; a.out_f32 = out_f32; a.out_stride = ostride; a.act_tanh = act;
    rnn_gemm_wmma<<<dim3(N / 128), dim3(128), 0, stream>>>(a);
  };

  float* out  = (float*)d_out;
  float* o4e  = out;
  float* o4i  = o4e  + (size_t)B * T * O4E;
  float* o23e = o4i  + (size_t)B * T * O4I;
  float* o23i = o23e + (size_t)B * T * O23E;

  const long long xs = (long long)T * XON;  // batch stride of x at fixed t

  for (int t = 0; t < T; ++t) {
    const int p = t & 1, q = p ^ 1;     // read old = [p], write new = [q]
    const unsigned short* xon_t  = xon_bf  + (long long)t * XON;
    const unsigned short* xoff_t = xoff_bf + (long long)t * XOFF;

    // h4e_new = tanh([x, h4i_old] @ Wih^T + h4e_old @ Whh^T + b)
    gemm(H4E, { S(xon_t,  xs,  w4e_ih,            IN + H4I, XON),
                S(xoff_t, xs,  w4e_ih + XON,      IN + H4I, XOFF),
                S(hb4i[p], H4I, w4e_ih + IN,      IN + H4I, H4I),
                S(hb4e[p], H4E, w4e_hh,           H4E,      H4E) },
         bs4e, hb4e[q], nullptr, 0, 1);
    // h4i_new uses h4e_NEW
    gemm(H4I, { S(xon_t,  xs,  w4i_ih,            IN + H4E, XON),
                S(xoff_t, xs,  w4i_ih + XON,      IN + H4E, XOFF),
                S(hb4e[q], H4E, w4i_ih + IN,      IN + H4E, H4E),
                S(hb4i[p], H4I, w4i_hh,           H4I,      H4I) },
         bs4i, hb4i[q], nullptr, 0, 1);
    // FC heads for L4 (fp32 out, no activation)
    gemm(O4E, { S(hb4e[q], H4E, wfc4e, H4E, H4E) }, bfc4e,
         nullptr, o4e + (long long)t * O4E, (long long)T * O4E, 0);
    gemm(O4I, { S(hb4i[q], H4I, wfc4i, H4I, H4I) }, bfc4i,
         nullptr, o4i + (long long)t * O4I, (long long)T * O4I, 0);
    // h23e_new = tanh([h4i_new, h4e_new, h23i_old] @ Wih^T + h23e_old @ Whh^T + b)
    gemm(H23E, { S(hb4i[q],  H4I,  w23e_ih,              H4E + H4I + H23I, H4I),
                 S(hb4e[q],  H4E,  w23e_ih + H4I,        H4E + H4I + H23I, H4E),
                 S(hb23i[p], H23I, w23e_ih + H4I + H4E,  H4E + H4I + H23I, H23I),
                 S(hb23e[p], H23E, w23e_hh,              H23E,             H23E) },
         bs23e, hb23e[q], nullptr, 0, 1);
    // h23i_new uses h23e_NEW
    gemm(H23I, { S(hb4i[q],  H4I,  w23i_ih,              H4E + H4I + H23E, H4I),
                 S(hb4e[q],  H4E,  w23i_ih + H4I,        H4E + H4I + H23E, H4E),
                 S(hb23e[q], H23E, w23i_ih + H4I + H4E,  H4E + H4I + H23E, H23E),
                 S(hb23i[p], H23I, w23i_hh,              H23I,             H23I) },
         bs23i, hb23i[q], nullptr, 0, 1);
    // FC heads for L23
    gemm(O23E, { S(hb23e[q], H23E, wfc23e, H23E, H23E) }, bfc23e,
         nullptr, o23e + (long long)t * O23E, (long long)T * O23E, 0);
    gemm(O23I, { S(hb23i[q], H23I, wfc23i, H23I, H23I) }, bfc23i,
         nullptr, o23i + (long long)t * O23I, (long long)T * O23I, 0);
  }
}
